// CausalSelfAttention_20718922236087
// MI455X (gfx1250) — matrix operand order repaired and checked by reading, not compiled, not run on hardware
//
#include <hip/hip_runtime.h>

#define B_  2
#define T_  2048
#define H_  16
#define D_  64
#define C_  1024
#define M_  (B_*T_)    // 4096 rows
#define HD_ (H_*D_)    // 1024

typedef __attribute__((ext_vector_type(16))) __bf16 v16bf;
typedef __attribute__((ext_vector_type(8)))  float  v8f;
typedef unsigned short u16;

// ---------- helpers ----------
static __device__ __forceinline__ u16 f2bf(float f) {
  union { float f; unsigned u; } c; c.f = f;
  unsigned r = c.u + 0x7FFFu + ((c.u >> 16) & 1u);   // round-to-nearest-even
  return (u16)(r >> 16);
}

// Load a 16-element bf16 fragment from two 16-byte chunks.
static __device__ __forceinline__ v16bf load_frag(const u16* p0, const u16* p1) {
  union { v16bf v; uint4 q[2]; } u;
  u.q[0] = *(const uint4*)p0;
  u.q[1] = *(const uint4*)p1;
  return u.v;
}

static __device__ __forceinline__ v8f wmma_bf16(v16bf a, v16bf b, v8f c) {
  // (neg_a, A, neg_b, B, c_mod, C, reuse_a, reuse_b)
  return __builtin_amdgcn_wmma_f32_16x16x32_bf16(false, a, false, b, (short)0, c,
                                                 false, false);
}

static __device__ __forceinline__ float red16_max(float v) {   // reduce within 16-lane half
  v = fmaxf(v, __shfl_xor(v, 1, 32));
  v = fmaxf(v, __shfl_xor(v, 2, 32));
  v = fmaxf(v, __shfl_xor(v, 4, 32));
  v = fmaxf(v, __shfl_xor(v, 8, 32));
  return v;
}
static __device__ __forceinline__ float red16_sum(float v) {
  v += __shfl_xor(v, 1, 32);
  v += __shfl_xor(v, 2, 32);
  v += __shfl_xor(v, 4, 32);
  v += __shfl_xor(v, 8, 32);
  return v;
}
static __device__ __forceinline__ float red32_sum(float v) {
  v += __shfl_xor(v, 16, 32);
  return red16_sum(v);
}
static __device__ __forceinline__ float softplus_f(float x) {
  return (x > 20.f) ? x : log1pf(__expf(x));
}

// Async copy of 16 bytes/lane from global to LDS (ASYNCcnt-tracked DMA path).
static __device__ __forceinline__ void async_g2l_b128(unsigned lds_off, const void* gaddr) {
  asm volatile("global_load_async_to_lds_b128 %0, %1, off"
               :: "v"(lds_off), "v"(gaddr) : "memory");
}
static __device__ __forceinline__ void wait_async0() {
  asm volatile("s_wait_asynccnt 0x0" ::: "memory");
}

// ---------- conversion kernels ----------
__global__ void k_f32_to_bf16(const float* __restrict__ in, u16* __restrict__ out, int n) {
  int i = blockIdx.x * blockDim.x + threadIdx.x;
  if (i < n) out[i] = f2bf(in[i]);
}

// W[rows,cols] (in,out) -> Wt[cols,rows] bf16  (B-operand wants [N,K] row-major)
__global__ void k_transpose_bf16(const float* __restrict__ W, u16* __restrict__ Wt,
                                 int rows, int cols) {
  int i = blockIdx.x * blockDim.x + threadIdx.x;
  if (i < rows * cols) {
    int k = i / cols, n = i % cols;
    Wt[(size_t)n * rows + k] = f2bf(W[i]);
  }
}

// ---------- bf16 WMMA GEMM: C[M,N] = A[M,K] * Bt[N,K]^T ----------
__global__ __launch_bounds__(256)
void k_gemm_bf16(const u16* __restrict__ A, const u16* __restrict__ Bt,
                 float* __restrict__ Cmat, int M, int N, int K) {
  const int l  = threadIdx.x & 31;
  const int w  = threadIdx.x >> 5;
  const int ln = l & 15;       // N (B/C/D) or M (A) position within half
  const int hs = l >> 4;       // lane half selector
  const int m0 = blockIdx.y * 128 + w * 16;
  const int n0 = blockIdx.x * 64;

  v8f acc[4] = {};
  const u16* pa_row = A + (size_t)(m0 + ln) * K;

  for (int kk = 0; kk < K; kk += 32) {
    // A frag: lanes 0-15 hold K=kk..kk+15; lanes 16-31 +16
    const u16* pa = pa_row + kk + 16 * hs;
    __builtin_prefetch(pa + 64, 0, 0);
    v16bf a = load_frag(pa, pa + 8);
#pragma unroll
    for (int nt = 0; nt < 4; ++nt) {
      // B frag: lane's column n, 16 contiguous K values (halves offset by 16)
      const u16* pb = Bt + (size_t)(n0 + nt * 16 + ln) * K + kk + 16 * hs;
      v16bf b = load_frag(pb, pb + 8);
      acc[nt] = wmma_bf16(a, b, acc[nt]);
    }
  }
  // C layout: VGPR r -> row m0+r (lanes 0-15) / m0+r+8 (lanes 16-31), col = ln
#pragma unroll
  for (int nt = 0; nt < 4; ++nt)
#pragma unroll
    for (int r = 0; r < 8; ++r)
      Cmat[(size_t)(m0 + r + 8 * hs) * N + n0 + nt * 16 + ln] = acc[nt][r];
}

// ---------- RMSNorm(q,k) + key-gate + bf16 pack; one wave per (row, head) ----------
__global__ __launch_bounds__(128)
void k_norm_gate(const float* __restrict__ qraw, const float* __restrict__ kraw,
                 const float* __restrict__ vraw, const float* __restrict__ qw,
                 const float* __restrict__ kw, const float* __restrict__ u,
                 u16* __restrict__ qbf, u16* __restrict__ kbf, u16* __restrict__ vbf,
                 float* __restrict__ gate) {
  const int l   = threadIdx.x & 31;
  const int w   = threadIdx.x >> 5;
  const int g   = blockIdx.x * 4 + w;          // 0..M_*H_-1
  const int h   = g & (H_ - 1);
  const int row = g >> 4;                      // b*T + t
  const int t   = row & (T_ - 1);
  const int b   = row >> 11;
  const size_t base = (size_t)row * HD_ + h * D_;
  const int d0 = l * 2;

  float q0 = qraw[base + d0], q1 = qraw[base + d0 + 1];
  float rq = rsqrtf(red32_sum(q0 * q0 + q1 * q1) * (1.f / D_) + 1e-5f);
  qbf[base + d0]     = f2bf(q0 * rq * qw[d0]);
  qbf[base + d0 + 1] = f2bf(q1 * rq * qw[d0 + 1]);

  float k0 = kraw[base + d0], k1 = kraw[base + d0 + 1];
  float rk = rsqrtf(red32_sum(k0 * k0 + k1 * k1) * (1.f / D_) + 1e-5f);
  float kn0 = k0 * rk * kw[d0], kn1 = k1 * rk * kw[d0 + 1];
  kbf[base + d0]     = f2bf(kn0);
  kbf[base + d0 + 1] = f2bf(kn1);

  float u0 = u[h * D_ + d0], u1 = u[h * D_ + d0 + 1];
  float nrm = fmaxf(sqrtf(red32_sum(u0 * u0 + u1 * u1)), 1e-6f);
  float dot = red32_sum(kn0 * u0 + kn1 * u1) / nrm;
  float gt  = softplus_f(dot * 0.125f);        // 1/sqrt(64)
  if (l == 0) gate[((size_t)(b * H_ + h)) * T_ + t] = gt;

  vbf[base + d0]     = f2bf(vraw[base + d0]);
  vbf[base + d0 + 1] = f2bf(vraw[base + d0 + 1]);
}

// ---------- flash attention with key-gated ALiBi bias; one wave per 16-query tile ----------
// Per-wave LDS: Vt[64][32] (transposed V), P[16][32], Krow[32][64] (async-staged K tile)
#define WSLC_ (D_*32 + 16*32 + 32*D_)
__global__ __launch_bounds__(128)
void k_flash(const u16* __restrict__ qbf, const u16* __restrict__ kbf,
             const u16* __restrict__ vbf, const float* __restrict__ gate,
             const float* __restrict__ omega, u16* __restrict__ ybf) {
  __shared__ u16 lds[4 * WSLC_];
  const int l  = threadIdx.x & 31;
  const int w  = threadIdx.x >> 5;
  const int ln = l & 15;
  const int hs = l >> 4;
  u16* Vt = lds + w * WSLC_;
  u16* Pl = Vt + D_ * 32;
  u16* Kl = Pl + 16 * 32;          // row-major K tile: [key 0..31][feat 0..63]

  const int g  = blockIdx.x * 4 + w;
  const int qt = g & 127;
  const int h  = (g >> 7) & 15;
  const int b  = g >> 11;
  const int qb = qt * 16;

  // omega_eff = softplus(omega[h]) * alibi_slope(h);  H=16 -> slope = 2^(-0.5*(h+1))
  const float ome = softplus_f(omega[h]) * exp2f(-0.5f * (float)(h + 1));

  // Q A-fragments (16x64 split into two K=32 chunks), held in registers
  const u16* qp = qbf + ((size_t)(b * T_) + qb + ln) * HD_ + h * D_;
  v16bf aQ[2];
#pragma unroll
  for (int c = 0; c < 2; ++c) {
    const u16* p = qp + c * 32 + 16 * hs;
    aQ[c] = load_frag(p, p + 8);
  }

  v8f O[4] = {};
  float mrow[8], lrow[8];
#pragma unroll
  for (int r = 0; r < 8; ++r) { mrow[r] = -1e30f; lrow[r] = 0.f; }

  const float* gp = gate + (size_t)(b * H_ + h) * T_;
  const unsigned kl_lane = (unsigned)(uintptr_t)(Kl + l * D_);  // LDS offset of lane's K row

  for (int kb = 0; kb < qb + 16; kb += 32) {
    // ---- async DMA: K tile (32 keys x 64 feat bf16, 4KB) global -> LDS row-major ----
    {
      const u16* krow = kbf + ((size_t)(b * T_) + kb + l) * HD_ + h * D_;
#pragma unroll
      for (int i = 0; i < 8; ++i)
        async_g2l_b128(kl_lane + i * 16, (const void*)(krow + i * 8));
    }

    // ---- stage V tile (32 keys x 64 feat) transposed into LDS: Vt[d][key] ----
    {
      const uint4* vp4 = (const uint4*)(vbf + ((size_t)(b * T_) + kb + l) * HD_ + h * D_);
#pragma unroll
      for (int i = 0; i < 8; ++i) {
        uint4 q = vp4[i];
        unsigned vv[4] = { q.x, q.y, q.z, q.w };
#pragma unroll
        for (int j = 0; j < 4; ++j) {
          Vt[(i * 8 + j * 2 + 0) * 32 + l] = (u16)(vv[j] & 0xffffu);
          Vt[(i * 8 + j * 2 + 1) * 32 + l] = (u16)(vv[j] >> 16);
        }
      }
    }

    wait_async0();   // K tile resident in LDS

    // ---- S = Q * K^T for 32 keys (two 16-key N tiles, K=64 in two chunks) ----
    v8f s0 = {}, s1 = {};
#pragma unroll
    for (int c = 0; c < 2; ++c) {
      const u16* kp0 = Kl + ln * D_ + c * 32 + 16 * hs;          // keys kb+ln
      s0 = wmma_bf16(aQ[c], load_frag(kp0, kp0 + 8), s0);
      const u16* kp1 = Kl + (16 + ln) * D_ + c * 32 + 16 * hs;   // keys kb+16+ln
      s1 = wmma_bf16(aQ[c], load_frag(kp1, kp1 + 8), s1);
    }

    const float g0 = gp[kb + ln] * ome;        // key-gated slope for column j0
    const float g1 = gp[kb + 16 + ln] * ome;
    const int j0 = kb + ln, j1 = kb + 16 + ln;

    // ---- online softmax (row r lives in 16 lanes of VGPR r; halves = rows r / r+8) ----
    float p0[8], p1[8], alpha[8];
#pragma unroll
    for (int r = 0; r < 8; ++r) {
      const int i = qb + r + 8 * hs;
      float x0 = (i >= j0) ? s0[r] * 0.125f - (float)(i - j0) * g0 : -1e30f;
      float x1 = (i >= j1) ? s1[r] * 0.125f - (float)(i - j1) * g1 : -1e30f;
      float mt = red16_max(fmaxf(x0, x1));
      float mn = fmaxf(mrow[r], mt);
      float al = __expf(mrow[r] - mn);
      float e0 = __expf(x0 - mn);
      float e1 = __expf(x1 - mn);
      lrow[r] = lrow[r] * al + red16_sum(e0 + e1);
      mrow[r] = mn; alpha[r] = al; p0[r] = e0; p1[r] = e1;
    }
#pragma unroll
    for (int nt = 0; nt < 4; ++nt)
#pragma unroll
      for (int r = 0; r < 8; ++r)
        O[nt][r] *= alpha[r];

    // ---- P (C-layout) -> LDS -> A-layout fragment ----
#pragma unroll
    for (int r = 0; r < 8; ++r) {
      Pl[(r + 8 * hs) * 32 + ln]      = f2bf(p0[r]);
      Pl[(r + 8 * hs) * 32 + 16 + ln] = f2bf(p1[r]);
    }
    asm volatile("s_wait_dscnt 0x0" ::: "memory");
    const u16* pp = Pl + ln * 32 + 16 * hs;
    v16bf aP = load_frag(pp, pp + 8);

    // ---- O += P * V (four N tiles of 16 features) ----
#pragma unroll
    for (int nt = 0; nt < 4; ++nt) {
      const u16* vp = Vt + (nt * 16 + ln) * 32 + 16 * hs;
      O[nt] = wmma_bf16(aP, load_frag(vp, vp + 8), O[nt]);
    }
  }

  // ---- epilogue: normalize and store bf16 y ----
  float inv[8];
#pragma unroll
  for (int r = 0; r < 8; ++r) inv[r] = 1.0f / lrow[r];
#pragma unroll
  for (int nt = 0; nt < 4; ++nt)
#pragma unroll
    for (int r = 0; r < 8; ++r)
      ybf[((size_t)(b * T_) + qb + r + 8 * hs) * HD_ + h * D_ + nt * 16 + ln] =
          f2bf(O[nt][r] * inv[r]);
}

// ---------- host launch ----------
extern "C" void kernel_launch(void* const* d_in, const int* in_sizes, int n_in,
                              void* d_out, int out_size, void* d_ws, size_t ws_size,
                              hipStream_t stream) {
  (void)in_sizes; (void)n_in; (void)out_size; (void)ws_size;
  const float* x     = (const float*)d_in[0];
  const float* Wq    = (const float*)d_in[1];
  const float* Wk    = (const float*)d_in[2];
  const float* Wv    = (const float*)d_in[3];
  const float* Wp    = (const float*)d_in[4];
  const float* qw    = (const float*)d_in[5];
  const float* kw    = (const float*)d_in[6];
  const float* omega = (const float*)d_in[7];
  const float* u     = (const float*)d_in[8];

  char* ws = (char*)d_ws;
  const size_t MB = 1024ull * 1024ull;
  u16*   xbf   = (u16*)(ws);             //  8 MB: x bf16 [4096,1024]
  u16*   wqT   = (u16*)(ws +  8 * MB);   //  2 MB each, transposed [N,K]
  u16*   wkT   = (u16*)(ws + 10 * MB);
  u16*   wvT   = (u16*)(ws + 12 * MB);
  u16*   wpT   = (u16*)(ws + 14 * MB);
  float* qraw  = (float*)(ws + 16 * MB); // 16 MB each
  float* kraw  = (float*)(ws + 32 * MB);
  float* vraw  = (float*)(ws + 48 * MB);
  u16*   qbf   = (u16*)(ws + 64 * MB);   //  8 MB each
  u16*   kbf   = (u16*)(ws + 72 * MB);
  u16*   vbf   = (u16*)(ws + 80 * MB);
  u16*   ybf   = (u16*)(ws + 88 * MB);
  float* gateb = (float*)(ws + 96 * MB); // 256 KB

  const int CB = 256;
  k_f32_to_bf16<<<(M_ * C_ + CB - 1) / CB, CB, 0, stream>>>(x, xbf, M_ * C_);
  k_transpose_bf16<<<(C_ * HD_ + CB - 1) / CB, CB, 0, stream>>>(Wq, wqT, C_, HD_);
  k_transpose_bf16<<<(C_ * HD_ + CB - 1) / CB, CB, 0, stream>>>(Wk, wkT, C_, HD_);
  k_transpose_bf16<<<(C_ * HD_ + CB - 1) / CB, CB, 0, stream>>>(Wv, wvT, C_, HD_);
  k_transpose_bf16<<<(HD_ * C_ + CB - 1) / CB, CB, 0, stream>>>(Wp, wpT, HD_, C_);

  dim3 gg(HD_ / 64, M_ / 128);
  k_gemm_bf16<<<gg, 256, 0, stream>>>(xbf, wqT, qraw, M_, HD_, C_);
  k_gemm_bf16<<<gg, 256, 0, stream>>>(xbf, wkT, kraw, M_, HD_, C_);
  k_gemm_bf16<<<gg, 256, 0, stream>>>(xbf, wvT, vraw, M_, HD_, C_);

  k_norm_gate<<<(M_ * H_) / 4, 128, 0, stream>>>(qraw, kraw, vraw, qw, kw, u,
                                                 qbf, kbf, vbf, gateb);

  k_flash<<<(B_ * H_ * (T_ / 16)) / 4, 128, 0, stream>>>(qbf, kbf, vbf, gateb,
                                                         omega, ybf);

  dim3 g2(C_ / 64, M_ / 128);
  k_gemm_bf16<<<g2, 256, 0, stream>>>(ybf, wpT, (float*)d_out, M_, C_, HD_);
}
